// MultiHeadAttention_20091857010773
// MI455X (gfx1250) — matrix operand order repaired and checked by reading, not compiled, not run on hardware
//
#include <hip/hip_runtime.h>

#define B_  2
#define T_  2048
#define D_  1024
#define H_  16
#define HD_ 64
#define BT_ (B_*T_)

typedef __attribute__((ext_vector_type(16))) __bf16       v16bf;
typedef __attribute__((ext_vector_type(8)))  float        v8f;
typedef __attribute__((ext_vector_type(4)))  unsigned int u32x4;

#if __has_builtin(__builtin_amdgcn_global_load_async_to_lds_b128) && \
    __has_builtin(__builtin_amdgcn_s_wait_asynccnt)
#define HAVE_ASYNC_LDS 1
#endif

#ifdef HAVE_ASYNC_LDS
typedef __attribute__((ext_vector_type(4))) int i32x4_;
// 16B global -> LDS, per-lane addresses, tracked by ASYNCcnt.
// (generic LDS addr low 32 bits == LDS offset on CDNA5; global flat == global)
__device__ __forceinline__ void async_b128(const unsigned short* g, unsigned short* l) {
  __builtin_amdgcn_global_load_async_to_lds_b128(
      (__attribute__((address_space(1))) i32x4_*)(unsigned long long)(uintptr_t)g,
      (__attribute__((address_space(3))) i32x4_*)(unsigned)(uintptr_t)l,
      0, 0);
}
#endif

__device__ __forceinline__ unsigned short f2bf_us(float f) {
  unsigned u = __builtin_bit_cast(unsigned, f);
  unsigned r = u + 0x7FFFu + ((u >> 16) & 1u);   // round-to-nearest-even
  return (unsigned short)(r >> 16);
}

__device__ __forceinline__ v16bf frag_pack(u32x4 lo, u32x4 hi) {
  union { u32x4 q[2]; v16bf f; } u;
  u.q[0] = lo; u.q[1] = hi;
  return u.f;
}

// Butterfly reductions over each 16-lane half of a wave32 via ds_swizzle.
__device__ __forceinline__ float red16_max(float v) {
  v = fmaxf(v, __builtin_bit_cast(float, __builtin_amdgcn_ds_swizzle(__builtin_bit_cast(int, v), 0x041f)));
  v = fmaxf(v, __builtin_bit_cast(float, __builtin_amdgcn_ds_swizzle(__builtin_bit_cast(int, v), 0x081f)));
  v = fmaxf(v, __builtin_bit_cast(float, __builtin_amdgcn_ds_swizzle(__builtin_bit_cast(int, v), 0x101f)));
  v = fmaxf(v, __builtin_bit_cast(float, __builtin_amdgcn_ds_swizzle(__builtin_bit_cast(int, v), 0x201f)));
  return v;
}
__device__ __forceinline__ float red16_sum(float v) {
  v += __builtin_bit_cast(float, __builtin_amdgcn_ds_swizzle(__builtin_bit_cast(int, v), 0x041f));
  v += __builtin_bit_cast(float, __builtin_amdgcn_ds_swizzle(__builtin_bit_cast(int, v), 0x081f));
  v += __builtin_bit_cast(float, __builtin_amdgcn_ds_swizzle(__builtin_bit_cast(int, v), 0x101f));
  v += __builtin_bit_cast(float, __builtin_amdgcn_ds_swizzle(__builtin_bit_cast(int, v), 0x201f));
  return v;
}

// ---- conversions ------------------------------------------------------------
__global__ void f32_to_bf16_kernel(const float* __restrict__ in,
                                   unsigned short* __restrict__ out, int n) {
  int i = blockIdx.x * blockDim.x + threadIdx.x;
  if (i < n) out[i] = f2bf_us(in[i]);
}
// W [K][N] fp32 -> Wt [N][K] bf16
__global__ void f32_to_bf16_T(const float* __restrict__ in,
                              unsigned short* __restrict__ out, int Kd, int Nd) {
  int i = blockIdx.x * blockDim.x + threadIdx.x;
  if (i < Kd * Nd) {
    int n = i / Kd, k = i - n * Kd;
    out[i] = f2bf_us(in[(size_t)k * Nd + n]);
  }
}

// ---- tiled bf16 GEMM: C[M,N] = A[M,K] * Wt[N,K]^T ---------------------------
#define AS_S 40   // 32 + 8 pad ushorts (80B rows, 16B aligned)
#define BS_S 40

// OUT_MODE: 0 = bf16 [M][N] (*scale), 1 = f32 [M][N] + bias, 2 = bf16 V-transposed [B*H][HD][T]
template<int OUT_MODE>
__global__ __launch_bounds__(128) void gemm_bf16(
    const unsigned short* __restrict__ A,
    const unsigned short* __restrict__ Wt,
    void* __restrict__ Cout,
    const float* __restrict__ bias,
    int M, int N, int K, float scale) {
  __shared__ unsigned short As[2][128 * AS_S];
  __shared__ unsigned short Bp[2][64 * BS_S];

  const int bm = blockIdx.y, bn = blockIdx.x;
  const int tid  = threadIdx.x;
  const int wave = tid >> 5;
  const int lane = tid & 31;
  const int lh   = lane >> 4;
  const int col  = lane & 15;

  v8f acc[2][4] = {};

  const int ar  = tid >> 2;                                  // 0..31
  const int ac  = tid & 3;                                   // k-chunk (8 bf16)
  const int apc = ac;                                        // contiguous chunk
  const int br  = tid >> 1;                                  // 0..63 (n row)
  const int bc  = (tid & 1) * 16;                            // k offset

  const unsigned short* Abase = A  + (size_t)(bm*128 + ar) * K + ac*8;
  const unsigned short* Wbase = Wt + (size_t)(bn*64  + br) * K + bc;

#ifdef HAVE_ASYNC_LDS
  auto stage_async = [&](int buf, int k1) {
    async_b128(Abase + k1,                &As[buf][(ar     ) * AS_S + apc*8]);
    async_b128(Abase + (size_t)32*K + k1, &As[buf][(ar + 32) * AS_S + apc*8]);
    async_b128(Abase + (size_t)64*K + k1, &As[buf][(ar + 64) * AS_S + apc*8]);
    async_b128(Abase + (size_t)96*K + k1, &As[buf][(ar + 96) * AS_S + apc*8]);
    async_b128(Wbase + k1,                &Bp[buf][br * BS_S + bc]);
    async_b128(Wbase + k1 + 8,            &Bp[buf][br * BS_S + bc + 8]);
  };
  stage_async(0, 0);
#else
  u32x4 a0, a1, a2, a3, b0, b1;
  a0 = *(const u32x4*)(Abase);
  a1 = *(const u32x4*)(Abase + (size_t)32 * K);
  a2 = *(const u32x4*)(Abase + (size_t)64 * K);
  a3 = *(const u32x4*)(Abase + (size_t)96 * K);
  b0 = *(const u32x4*)(Wbase);
  b1 = *(const u32x4*)(Wbase + 8);
  *(u32x4*)&As[0][(ar     ) * AS_S + apc*8] = a0;
  *(u32x4*)&As[0][(ar + 32) * AS_S + apc*8] = a1;
  *(u32x4*)&As[0][(ar + 64) * AS_S + apc*8] = a2;
  *(u32x4*)&As[0][(ar + 96) * AS_S + apc*8] = a3;
  *(u32x4*)&Bp[0][br * BS_S + bc]     = b0;
  *(u32x4*)&Bp[0][br * BS_S + bc + 8] = b1;
#endif

  const int steps = K >> 5;
  for (int i = 0; i < steps; ++i) {
#ifdef HAVE_ASYNC_LDS
    __builtin_amdgcn_s_wait_asynccnt(0);   // our fills of buffer i&1 are done
#endif
    __syncthreads();
    if (i + 1 < steps) {
      const int k1 = (i + 1) * 32;
#ifdef HAVE_ASYNC_LDS
      stage_async((i + 1) & 1, k1);
#else
      a0 = *(const u32x4*)(Abase + k1);
      a1 = *(const u32x4*)(Abase + (size_t)32 * K + k1);
      a2 = *(const u32x4*)(Abase + (size_t)64 * K + k1);
      a3 = *(const u32x4*)(Abase + (size_t)96 * K + k1);
      b0 = *(const u32x4*)(Wbase + k1);
      b1 = *(const u32x4*)(Wbase + k1 + 8);
#endif
      if (i + 2 < steps) {   // -> global_prefetch_b8
        __builtin_prefetch(Abase + k1 + 32, 0, 3);
        __builtin_prefetch(Wbase + k1 + 32, 0, 3);
      }
    }
    const unsigned short* as = As[i & 1];
    const unsigned short* bp = Bp[i & 1];
    v16bf af[2], bf[4];
#pragma unroll
    for (int mt = 0; mt < 2; ++mt) {
      const unsigned short* ap = &as[(wave*32 + mt*16 + (lane & 15)) * AS_S + lh*16];
      af[mt] = frag_pack(*(const u32x4*)ap, *(const u32x4*)(ap + 8));
    }
#pragma unroll
    for (int nt = 0; nt < 4; ++nt) {
      const unsigned short* bq = &bp[(nt*16 + col) * BS_S + lh*16];
      bf[nt] = frag_pack(*(const u32x4*)bq, *(const u32x4*)(bq + 8));
    }
#pragma unroll
    for (int nt = 0; nt < 4; ++nt)
#pragma unroll
      for (int mt = 0; mt < 2; ++mt)
        acc[mt][nt] = __builtin_amdgcn_wmma_f32_16x16x32_bf16(
            false, af[mt], false, bf[nt], (short)0, acc[mt][nt], false, false);

#ifndef HAVE_ASYNC_LDS
    if (i + 1 < steps) {
      unsigned short* asn = As[(i + 1) & 1];
      unsigned short* bpn = Bp[(i + 1) & 1];
      *(u32x4*)&asn[(ar     ) * AS_S + apc*8] = a0;
      *(u32x4*)&asn[(ar + 32) * AS_S + apc*8] = a1;
      *(u32x4*)&asn[(ar + 64) * AS_S + apc*8] = a2;
      *(u32x4*)&asn[(ar + 96) * AS_S + apc*8] = a3;
      *(u32x4*)&bpn[br * BS_S + bc]     = b0;
      *(u32x4*)&bpn[br * BS_S + bc + 8] = b1;
    }
#endif
  }

#pragma unroll
  for (int mt = 0; mt < 2; ++mt) {
#pragma unroll
    for (int nt = 0; nt < 4; ++nt) {
#pragma unroll
      for (int r = 0; r < 8; ++r) {
        const int m = bm*128 + wave*32 + mt*16 + lh*8 + r;
        const int n = bn*64 + nt*16 + col;
        float v = acc[mt][nt][r] * scale;
        if (OUT_MODE == 0) {
          ((unsigned short*)Cout)[(size_t)m * N + n] = f2bf_us(v);
        } else if (OUT_MODE == 1) {
          ((float*)Cout)[(size_t)m * N + n] = v + bias[n];
        } else {
          const int bb = m >> 11, t = m & (T_ - 1);     // M == B_*T_
          const int hh = n >> 6,  d = n & (HD_ - 1);    // N == D_
          ((unsigned short*)Cout)[((size_t)(bb*H_ + hh)*HD_ + d)*T_ + t] = f2bf_us(v);
        }
      }
    }
  }
}

// ---- flash attention --------------------------------------------------------
#define KS_S 72   // 64 + 8
#define VT_S 40   // 32 + 8
#define PS_S 40

__global__ __launch_bounds__(128) void attn_fwd(
    const unsigned short* __restrict__ Q,    // [BT][D] bf16 (pre-scaled by 0.125)
    const unsigned short* __restrict__ K,    // [BT][D] bf16
    const unsigned short* __restrict__ Vt,   // [B*H][HD][T] bf16
    unsigned short* __restrict__ CTX) {      // [BT][D] bf16
  __shared__ unsigned short Ks[2][32 * KS_S];
  __shared__ unsigned short Vs[2][64 * VT_S];
  __shared__ unsigned short Ps[4][16 * PS_S];

  const int qblock = blockIdx.x & 31;   // T/64
  const int bh = blockIdx.x >> 5;       // b*H + h
  const int b  = bh >> 4;
  const int h  = bh & 15;

  const int tid  = threadIdx.x;
  const int wave = tid >> 5;
  const int lane = tid & 31;
  const int lh   = lane >> 4;
  const int col  = lane & 15;
  const int qt0  = qblock * 64 + wave * 16;

  // Q A-fragments (d 0..31, 32..63), gathered once from global
  v16bf qa0, qa1;
  {
    const unsigned short* qb = Q + (size_t)(b * T_ + qt0) * D_ + h * HD_;
    const int row = lane & 15;
#pragma unroll
    for (int j = 0; j < 8; ++j) {
      const int kidx = lh * 16 + j * 2;
      unsigned u0 = *(const unsigned*)(qb + (size_t)row * D_ + kidx);
      unsigned u1 = *(const unsigned*)(qb + (size_t)row * D_ + 32 + kidx);
      qa0[2*j]   = __builtin_bit_cast(__bf16, (unsigned short)(u0 & 0xFFFFu));
      qa0[2*j+1] = __builtin_bit_cast(__bf16, (unsigned short)(u0 >> 16));
      qa1[2*j]   = __builtin_bit_cast(__bf16, (unsigned short)(u1 & 0xFFFFu));
      qa1[2*j+1] = __builtin_bit_cast(__bf16, (unsigned short)(u1 >> 16));
    }
  }

  float mrow[8], lrow[8];
#pragma unroll
  for (int r = 0; r < 8; ++r) { mrow[r] = -1e30f; lrow[r] = 0.0f; }
  v8f acc[4] = {};

  const unsigned short* Kbase = K + (size_t)(b * T_) * D_ + h * HD_;
  const unsigned short* Vtb   = Vt + (size_t)bh * HD_ * T_;
  const int key = tid >> 2;            // staging: K row
  const int ch  = (tid & 3) * 16;      // staging: K d-chunk
  const int dr  = tid >> 1;            // staging: Vt row (d)
  const int kc  = (tid & 1) * 16;      // staging: Vt key-chunk

#ifdef HAVE_ASYNC_LDS
  auto stage_async = [&](int buf, int kb2) {
    const unsigned short* kp = Kbase + (size_t)(kb2 + key) * D_ + ch;
    async_b128(kp,     &Ks[buf][key * KS_S + ch]);
    async_b128(kp + 8, &Ks[buf][key * KS_S + ch + 8]);
    const unsigned short* vp = Vtb + (size_t)dr * T_ + kb2 + kc;
    async_b128(vp,     &Vs[buf][dr * VT_S + kc]);
    async_b128(vp + 8, &Vs[buf][dr * VT_S + kc + 8]);
  };
  stage_async(0, 0);
#else
  u32x4 kk0, kk1, vv0, vv1;
  {
    const unsigned short* kp = Kbase + (size_t)key * D_ + ch;
    kk0 = *(const u32x4*)kp;
    kk1 = *(const u32x4*)(kp + 8);
    const unsigned short* vp = Vtb + (size_t)dr * T_ + kc;
    vv0 = *(const u32x4*)vp;
    vv1 = *(const u32x4*)(vp + 8);
    *(u32x4*)&Ks[0][key * KS_S + ch]     = kk0;
    *(u32x4*)&Ks[0][key * KS_S + ch + 8] = kk1;
    *(u32x4*)&Vs[0][dr * VT_S + kc]      = vv0;
    *(u32x4*)&Vs[0][dr * VT_S + kc + 8]  = vv1;
  }
#endif

  const int nkb = qblock * 2 + 2;
  for (int ib = 0; ib < nkb; ++ib) {
    const int kb = ib * 32;
#ifdef HAVE_ASYNC_LDS
    __builtin_amdgcn_s_wait_asynccnt(0);
#endif
    __syncthreads();
    if (ib + 1 < nkb) {
#ifdef HAVE_ASYNC_LDS
      stage_async((ib + 1) & 1, kb + 32);
#else
      const unsigned short* kp = Kbase + (size_t)(kb + 32 + key) * D_ + ch;
      kk0 = *(const u32x4*)kp;
      kk1 = *(const u32x4*)(kp + 8);
      const unsigned short* vp = Vtb + (size_t)dr * T_ + kb + 32 + kc;
      vv0 = *(const u32x4*)vp;
      vv1 = *(const u32x4*)(vp + 8);
#endif
      if (ib + 2 < nkb) {
        __builtin_prefetch(Kbase + (size_t)(kb + 64 + key) * D_ + ch, 0, 3);
        __builtin_prefetch(Vtb + (size_t)dr * T_ + kb + 64 + kc, 0, 3);
      }
    }

    if (kb <= qt0 + 15) {    // wave-uniform causal skip
      const unsigned short* ks = Ks[ib & 1];
      const unsigned short* vs = Vs[ib & 1];
      v8f s0 = {}, s1 = {};
      {
        const unsigned short* kr0 = &ks[(col     ) * KS_S + lh*16];
        const unsigned short* kr1 = &ks[(16 + col) * KS_S + lh*16];
        v16bf b00 = frag_pack(*(const u32x4*)kr0,        *(const u32x4*)(kr0 + 8));
        v16bf b01 = frag_pack(*(const u32x4*)(kr0 + 32), *(const u32x4*)(kr0 + 40));
        v16bf b10 = frag_pack(*(const u32x4*)kr1,        *(const u32x4*)(kr1 + 8));
        v16bf b11 = frag_pack(*(const u32x4*)(kr1 + 32), *(const u32x4*)(kr1 + 40));
        s0 = __builtin_amdgcn_wmma_f32_16x16x32_bf16(false, qa0, false, b00, (short)0, s0, false, false);
        s0 = __builtin_amdgcn_wmma_f32_16x16x32_bf16(false, qa1, false, b01, (short)0, s0, false, false);
        s1 = __builtin_amdgcn_wmma_f32_16x16x32_bf16(false, qa0, false, b10, (short)0, s1, false, false);
        s1 = __builtin_amdgcn_wmma_f32_16x16x32_bf16(false, qa1, false, b11, (short)0, s1, false, false);
      }
      if (kb + 31 > qt0) {   // mask only diagonal-overlapping blocks
#pragma unroll
        for (int r = 0; r < 8; ++r) {
          const int q = qt0 + lh * 8 + r;
          if (kb + col > q)      s0[r] = -1e30f;
          if (kb + 16 + col > q) s1[r] = -1e30f;
        }
      }
      // online softmax
      float alpha[8];
#pragma unroll
      for (int r = 0; r < 8; ++r) {
        const float mx = red16_max(fmaxf(s0[r], s1[r]));
        const float mn = fmaxf(mrow[r], mx);
        alpha[r] = __expf(mrow[r] - mn);
        mrow[r] = mn;
        const float p0 = __expf(s0[r] - mn);
        const float p1 = __expf(s1[r] - mn);
        s0[r] = p0; s1[r] = p1;
        lrow[r] = lrow[r] * alpha[r] + red16_sum(p0 + p1);
      }
#pragma unroll
      for (int nt = 0; nt < 4; ++nt)
#pragma unroll
        for (int r = 0; r < 8; ++r) acc[nt][r] *= alpha[r];

      // P (C-layout) -> per-wave LDS with contiguous k positions -> b128 A-frag
      unsigned short* Pw = &Ps[wave][0];
      const int p0pos = col;
      const int p1pos = col + 16;
#pragma unroll
      for (int r = 0; r < 8; ++r) {
        const int m = lh * 8 + r;
        Pw[m * PS_S + p0pos] = f2bf_us(s0[r]);
        Pw[m * PS_S + p1pos] = f2bf_us(s1[r]);
      }
      const unsigned short* pp = &Ps[wave][(lane & 15) * PS_S + lh*16];
      v16bf pa = frag_pack(*(const u32x4*)pp, *(const u32x4*)(pp + 8));
      v16bf bv[4];
#pragma unroll
      for (int nt = 0; nt < 4; ++nt) {
        const unsigned short* vr = &vs[(nt*16 + col) * VT_S + lh*16];
        bv[nt] = frag_pack(*(const u32x4*)vr, *(const u32x4*)(vr + 8));
      }
#pragma unroll
      for (int nt = 0; nt < 4; ++nt)
        acc[nt] = __builtin_amdgcn_wmma_f32_16x16x32_bf16(
            false, pa, false, bv[nt], (short)0, acc[nt], false, false);
    }

#ifndef HAVE_ASYNC_LDS
    if (ib + 1 < nkb) {
      unsigned short* ksn = Ks[(ib + 1) & 1];
      unsigned short* vsn = Vs[(ib + 1) & 1];
      *(u32x4*)&ksn[key * KS_S + ch]     = kk0;
      *(u32x4*)&ksn[key * KS_S + ch + 8] = kk1;
      *(u32x4*)&vsn[dr * VT_S + kc]      = vv0;
      *(u32x4*)&vsn[dr * VT_S + kc + 8]  = vv1;
    }
#endif
  }

  // normalize + write ctx (bf16)
#pragma unroll
  for (int r = 0; r < 8; ++r) lrow[r] = 1.0f / lrow[r];
#pragma unroll
  for (int nt = 0; nt < 4; ++nt) {
#pragma unroll
    for (int r = 0; r < 8; ++r) {
      const int q = qt0 + lh * 8 + r;
      CTX[(size_t)(b * T_ + q) * D_ + h * HD_ + nt * 16 + col] =
          f2bf_us(acc[nt][r] * lrow[r]);
    }
  }
}

// ---- host launch ------------------------------------------------------------
extern "C" void kernel_launch(void* const* d_in, const int* in_sizes, int n_in,
                              void* d_out, int out_size, void* d_ws, size_t ws_size,
                              hipStream_t stream) {
  const float* x  = (const float*)d_in[0];
  const float* wq = (const float*)d_in[1];
  const float* wk = (const float*)d_in[2];
  const float* wv = (const float*)d_in[3];
  const float* wo = (const float*)d_in[4];
  const float* bo = (const float*)d_in[5];

  char* ws = (char*)d_ws;
  size_t off = 0;
  auto take = [&](size_t bytes) -> char* {
    char* p = ws + off;
    off += (bytes + 255) & ~(size_t)255;
    return p;
  };
  unsigned short* xb  = (unsigned short*)take((size_t)BT_ * D_ * 2);
  unsigned short* wqt = (unsigned short*)take((size_t)D_ * D_ * 2);
  unsigned short* wkt = (unsigned short*)take((size_t)D_ * D_ * 2);
  unsigned short* wvt = (unsigned short*)take((size_t)D_ * D_ * 2);
  unsigned short* wot = (unsigned short*)take((size_t)D_ * D_ * 2);
  unsigned short* Qb  = (unsigned short*)take((size_t)BT_ * D_ * 2);
  unsigned short* Kb  = (unsigned short*)take((size_t)BT_ * D_ * 2);
  unsigned short* Vtb = (unsigned short*)take((size_t)BT_ * D_ * 2);
  unsigned short* Cb  = (unsigned short*)take((size_t)BT_ * D_ * 2);

  const int nx = BT_ * D_;
  const int nw = D_ * D_;
  f32_to_bf16_kernel<<<(nx + 255) / 256, 256, 0, stream>>>(x, xb, nx);
  f32_to_bf16_T<<<(nw + 255) / 256, 256, 0, stream>>>(wq, wqt, D_, D_);
  f32_to_bf16_T<<<(nw + 255) / 256, 256, 0, stream>>>(wk, wkt, D_, D_);
  f32_to_bf16_T<<<(nw + 255) / 256, 256, 0, stream>>>(wv, wvt, D_, D_);
  f32_to_bf16_T<<<(nw + 255) / 256, 256, 0, stream>>>(wo, wot, D_, D_);

  dim3 gg(D_ / 64, BT_ / 128);
  gemm_bf16<0><<<gg, 128, 0, stream>>>(xb, wqt, Qb,  nullptr, BT_, D_, D_, 0.125f);
  gemm_bf16<0><<<gg, 128, 0, stream>>>(xb, wkt, Kb,  nullptr, BT_, D_, D_, 1.0f);
  gemm_bf16<2><<<gg, 128, 0, stream>>>(xb, wvt, Vtb, nullptr, BT_, D_, D_, 1.0f);

  attn_fwd<<<B_ * H_ * (T_ / 64), 128, 0, stream>>>(Qb, Kb, Vtb, Cb);

  gemm_bf16<1><<<gg, 128, 0, stream>>>(Cb, wot, (float*)d_out, bo, BT_, D_, D_, 1.0f);
}
